// ProcessBlock_63677185131201
// MI455X (gfx1250) — matrix operand order repaired and checked by reading, not compiled, not run on hardware
//
#include <hip/hip_runtime.h>
#include <hip/hip_bf16.h>
#include <math.h>

// ---------------------------------------------------------------------------
// CDNA5 (gfx1250) implementation. All channel-mixing math (1x1 convs, dense
// k x k convs via implicit-GEMM/im2col, mamba projections, FFN, and DFT-as-GEMM
// for the FFC rfft2/irfft2) runs through a single WMMA GEMM kernel using
// v_wmma_f32_16x16x32_f16 (f16 in, f32 accumulate). Depthwise convs, Haar
// DWT/IDWT, layernorms and the sequential selective scan are VALU kernels.
//
// GEMM data path: A fragments gathered directly from global (each A-tile
// element is consumed by exactly one lane), B tile staged in LDS in *fragment
// order* (contiguous 32B v16h per lane -> ds_load_b128). The K loop is split
// into a branch-free interior loop (no bounds masks, pointer-walk addressing,
// float4 loads when the K stride is contiguous, speculative global_prefetch)
// and a masked tail loop for boundary / K-remainder / im2col tiles, so the
// accumulators never cross a branch seam (no per-iteration register copies)
// and EXEC is always all-ones at the WMMAs.
// ---------------------------------------------------------------------------

typedef __attribute__((ext_vector_type(16))) _Float16 v16h;
typedef __attribute__((ext_vector_type(8)))  float    v8f;

__device__ __forceinline__ float actf(float x, int act) {
  if (act == 1) return x > 0.f ? x : 0.f;                       // relu
  if (act == 2) {                                               // gelu (tanh)
    float t = tanhf(0.7978845608028654f * (x + 0.044715f * x * x * x));
    return 0.5f * x * (1.f + t);
  }
  if (act == 3) return x * (1.f / (1.f + __expf(-x)));          // silu
  if (act == 4) return (x > 20.f) ? x : log1pf(__expf(x));      // softplus
  return x;
}

// ---------------------------------------------------------------------------
// Batched strided GEMM, C[z](M,N) (+)= act(alpha * A[z](M,K) @ B[z](K,N) + bias)
// Optional im2col mode: B gathered from an NCHW image (zero-padded "same" conv).
// Tile: 64(M) x 64(N) x 32(K), 128 threads (4 waves), each wave computes a
// 16x64 strip with 4 wmma per K-step.
// ---------------------------------------------------------------------------
struct GemmP {
  const float* A; const float* B; const float* Xim; const float* bias; float* C;
  long as_m, as_k, batchA;
  long bs_k, bs_n, batchB;
  long cs_m, cs_n, batchC;
  long batchBias, imBatch;
  int M, N, K;
  float alpha;
  int act, acc, has_bias, im2col;
  int Cin, Hh, Ww, kh, kw;
};

__global__ void __launch_bounds__(128) k_gemm(GemmP p) {
  // B tile kept in fragment order: BsF[nt][lane][half]; one 32B v16h per slot.
  __shared__ __align__(32) _Float16 BsF[4][32][16];
  const int z = blockIdx.z;
  const int m0 = blockIdx.y * 64, n0 = blockIdx.x * 64;
  const int tid = threadIdx.x, wave = tid >> 5, lane = tid & 31;
  const float* Ap = p.A + (long)z * p.batchA;
  float* Cp = p.C + (long)z * p.batchC;
  v8f acc0 = {}, acc1 = {}, acc2 = {}, acc3 = {};

  // Wave-uniform full-tile flags.
  const bool mfull = (m0 + 64 <= p.M);
  const bool nfull = (n0 + 64 <= p.N);

  // A row handled by this lane (ISA 7.12.2 16-bit A layout):
  // lane 0..15 -> M rows, lanes 16..31 same rows / other K-halves.
  const int gm_a = m0 + (wave << 4) + (lane & 15);
  const bool mok = gm_a < p.M;
  const long abase = (long)(mok ? gm_a : (p.M - 1)) * p.as_m;
  const int khalf = (lane >> 4) << 4;          // +16 K offset for hi lanes
  const float* ap_it = Ap + abase;             // advances 32*as_k per K step

  // B fragment this thread stages: group snt, slot slane.
  const int snt = tid >> 5, slane = tid & 31;
  const int sgn = n0 + (snt << 4) + (slane & 15);
  const int skbase = (slane >> 4) << 4;        // K base 0 or 16 within tile
  const bool nok = sgn < p.N;
  const float* bp = p.im2col
      ? p.Xim
      : (p.B + (long)z * p.batchB +
         (long)(nok ? sgn : (p.N - 1)) * p.bs_n + (long)skbase * p.bs_k);

  const bool avec = (p.as_k == 1) && ((p.as_m & 3) == 0);
  const bool bvec = (p.bs_k == 1) && ((p.bs_n & 3) == 0);

  // Loop-invariant split: interior K steps handled mask-free.
  const int kfastN = (mfull && nfull && !p.im2col) ? (p.K & ~31) : 0;
  int k0 = 0;

  // ======================= interior K loop (branch-free) ===================
  for (; k0 < kfastN; k0 += 32) {
    union { v16h v; _Float16 h[16]; } ua, sb;
    if (avec) {
      float4 a0 = *(const float4*)(ap_it + khalf);
      float4 a1 = *(const float4*)(ap_it + khalf + 4);
      float4 a2 = *(const float4*)(ap_it + khalf + 8);
      float4 a3 = *(const float4*)(ap_it + khalf + 12);
      ua.h[0]=(_Float16)a0.x; ua.h[1]=(_Float16)a0.y; ua.h[2]=(_Float16)a0.z; ua.h[3]=(_Float16)a0.w;
      ua.h[4]=(_Float16)a1.x; ua.h[5]=(_Float16)a1.y; ua.h[6]=(_Float16)a1.z; ua.h[7]=(_Float16)a1.w;
      ua.h[8]=(_Float16)a2.x; ua.h[9]=(_Float16)a2.y; ua.h[10]=(_Float16)a2.z; ua.h[11]=(_Float16)a2.w;
      ua.h[12]=(_Float16)a3.x; ua.h[13]=(_Float16)a3.y; ua.h[14]=(_Float16)a3.z; ua.h[15]=(_Float16)a3.w;
    } else {
      const float* q = ap_it + (long)khalf * p.as_k;
#pragma unroll
      for (int i = 0; i < 8; i++) { ua.h[i] = (_Float16)(*q); q += p.as_k; }
      q = ap_it + (long)(8 + khalf) * p.as_k;
#pragma unroll
      for (int i = 8; i < 16; i++) { ua.h[i] = (_Float16)(*q); q += p.as_k; }
    }
    __builtin_prefetch(ap_it + 32 * p.as_k, 0, 3);   // speculative, safe OOB

    if (bvec) {
      float4 b0 = *(const float4*)(bp);
      float4 b1 = *(const float4*)(bp + 4);
      float4 b2 = *(const float4*)(bp + 8);
      float4 b3 = *(const float4*)(bp + 12);
      sb.h[0]=(_Float16)b0.x; sb.h[1]=(_Float16)b0.y; sb.h[2]=(_Float16)b0.z; sb.h[3]=(_Float16)b0.w;
      sb.h[4]=(_Float16)b1.x; sb.h[5]=(_Float16)b1.y; sb.h[6]=(_Float16)b1.z; sb.h[7]=(_Float16)b1.w;
      sb.h[8]=(_Float16)b2.x; sb.h[9]=(_Float16)b2.y; sb.h[10]=(_Float16)b2.z; sb.h[11]=(_Float16)b2.w;
      sb.h[12]=(_Float16)b3.x; sb.h[13]=(_Float16)b3.y; sb.h[14]=(_Float16)b3.z; sb.h[15]=(_Float16)b3.w;
    } else {
      const float* q = bp;
#pragma unroll
      for (int i = 0; i < 16; i++) { sb.h[i] = (_Float16)(*q); q += p.bs_k; }
    }
    __builtin_prefetch(bp + 32 * p.bs_k, 0, 3);

    __syncthreads();
    *(v16h*)&BsF[snt][slane][0] = sb.v;
    __syncthreads();
    v16h b0 = *(const v16h*)&BsF[0][lane][0];
    v16h b1 = *(const v16h*)&BsF[1][lane][0];
    v16h b2 = *(const v16h*)&BsF[2][lane][0];
    v16h b3 = *(const v16h*)&BsF[3][lane][0];
    acc0 = __builtin_amdgcn_wmma_f32_16x16x32_f16(false, ua.v, false, b0, (short)0, acc0, false, false);
    acc1 = __builtin_amdgcn_wmma_f32_16x16x32_f16(false, ua.v, false, b1, (short)0, acc1, false, false);
    acc2 = __builtin_amdgcn_wmma_f32_16x16x32_f16(false, ua.v, false, b2, (short)0, acc2, false, false);
    acc3 = __builtin_amdgcn_wmma_f32_16x16x32_f16(false, ua.v, false, b3, (short)0, acc3, false, false);
    ap_it += 32 * p.as_k;
    bp += 32 * p.bs_k;
  }

  // ============== tail / boundary / im2col K loop (masked) =================
  for (; k0 < p.K; k0 += 32) {
    union { v16h v; _Float16 h[16]; } ua, sb;
#pragma unroll
    for (int i = 0; i < 16; i++) {
      int gk = k0 + khalf + i;
      bool ok = mok && (gk < p.K);
      long o = abase + (long)(gk < p.K ? gk : 0) * p.as_k;
      float v = Ap[o];
      ua.h[i] = (_Float16)(ok ? v : 0.f);
    }
    if (!p.im2col) {
      const float* Bp = p.B + (long)z * p.batchB;
      long nb = (long)(nok ? sgn : (p.N - 1)) * p.bs_n;
#pragma unroll
      for (int i = 0; i < 16; i++) {
        int gk = k0 + skbase + i;
        bool ok = nok && (gk < p.K);
        float v = Bp[nb + (long)(gk < p.K ? gk : 0) * p.bs_k];
        sb.h[i] = (_Float16)(ok ? v : 0.f);
      }
    } else {
      const float* Xp = p.Xim + (long)z * p.imBatch;
      const int KK = p.kh * p.kw, ph = (p.kh - 1) >> 1, pw = (p.kw - 1) >> 1;
      int gnc = nok ? sgn : (p.N - 1);
      int h = gnc / p.Ww, w = gnc - h * p.Ww;
#pragma unroll
      for (int i = 0; i < 16; i++) {
        int gk = k0 + skbase + i;
        bool ok = nok && (gk < p.K);
        int gkc = (gk < p.K) ? gk : 0;
        int ci = gkc / KK, t = gkc - ci * KK;
        int ky = t / p.kw, kx = t - ky * p.kw;
        int hh = h + ky - ph, ww = w + kx - pw;
        bool inb = ok && ((unsigned)hh < (unsigned)p.Hh) &&
                         ((unsigned)ww < (unsigned)p.Ww);
        long addr = inb ? (((long)ci * p.Hh + hh) * p.Ww + ww) : 0;
        float v = Xp[addr];
        sb.h[i] = (_Float16)(inb ? v : 0.f);
      }
    }
    __syncthreads();
    *(v16h*)&BsF[snt][slane][0] = sb.v;
    __syncthreads();
    v16h b0 = *(const v16h*)&BsF[0][lane][0];
    v16h b1 = *(const v16h*)&BsF[1][lane][0];
    v16h b2 = *(const v16h*)&BsF[2][lane][0];
    v16h b3 = *(const v16h*)&BsF[3][lane][0];
    acc0 = __builtin_amdgcn_wmma_f32_16x16x32_f16(false, ua.v, false, b0, (short)0, acc0, false, false);
    acc1 = __builtin_amdgcn_wmma_f32_16x16x32_f16(false, ua.v, false, b1, (short)0, acc1, false, false);
    acc2 = __builtin_amdgcn_wmma_f32_16x16x32_f16(false, ua.v, false, b2, (short)0, acc2, false, false);
    acc3 = __builtin_amdgcn_wmma_f32_16x16x32_f16(false, ua.v, false, b3, (short)0, acc3, false, false);
  }

  // --- epilogue: alpha, bias, activation, optional accumulate ---
  const float* biasp = p.has_bias ? (p.bias + (long)z * p.batchBias) : nullptr;
  v8f accs[4] = { acc0, acc1, acc2, acc3 };
  for (int nt = 0; nt < 4; nt++) {
    int gn = n0 + (nt << 4) + (lane & 15);
    if (gn >= p.N) continue;
    union { v8f v; float f[8]; } uc; uc.v = accs[nt];
#pragma unroll
    for (int r = 0; r < 8; r++) {
      int gm = m0 + (wave << 4) + r + ((lane >> 4) << 3);
      if (gm >= p.M) continue;
      float v = p.alpha * uc.f[r];
      if (biasp) v += biasp[gm];
      v = actf(v, p.act);
      long o = (long)gm * p.cs_m + (long)gn * p.cs_n;
      if (p.acc) Cp[o] += v; else Cp[o] = v;
    }
  }
}

// ---------------------------------------------------------------------------
// Depthwise conv (groups == channels), NCHW with per-image strides.
// ---------------------------------------------------------------------------
__global__ void k_dwconv(const float* __restrict__ x, const float* __restrict__ w,
                         const float* __restrict__ bias, const float* __restrict__ scale,
                         float* __restrict__ y, int N, int Ch, int H, int W, int ks,
                         long in_img, long out_img, int act) {
  long idx = (long)blockIdx.x * blockDim.x + threadIdx.x;
  long total = (long)N * Ch * H * W;
  if (idx >= total) return;
  int wq = (int)(idx % W); long t = idx / W;
  int hq = (int)(t % H); t /= H;
  int c = (int)(t % Ch); int n = (int)(t / Ch);
  int p = (ks - 1) >> 1;
  const float* xp = x + (long)n * in_img + (long)c * H * W;
  const float* wp = w + (long)c * ks * ks;
  float acc = 0.f;
  for (int ky = 0; ky < ks; ky++) {
    int hh = hq + ky - p;
    if ((unsigned)hh >= (unsigned)H) continue;
    for (int kx = 0; kx < ks; kx++) {
      int ww = wq + kx - p;
      if ((unsigned)ww >= (unsigned)W) continue;
      acc += wp[ky * ks + kx] * xp[(long)hh * W + ww];
    }
  }
  if (bias) acc += bias[c];
  if (scale) acc *= scale[c];
  acc = actf(acc, act);
  y[(long)n * out_img + (long)c * H * W + (long)hq * W + wq] = acc;
}

// ---------------------------------------------------------------------------
// Haar DWT / IDWT (pointer-per-subband so channel- or batch-concat layouts work)
// ---------------------------------------------------------------------------
__global__ void k_dwt(const float* __restrict__ x, long x_img, int Cc, int H2, int W2,
                      float* ll, float* hl, float* lh, float* hh, long o_img, int P) {
  long idx = (long)blockIdx.x * blockDim.x + threadIdx.x;
  long total = (long)P * Cc * H2 * W2;
  if (idx >= total) return;
  int wq = (int)(idx % W2); long t = idx / W2;
  int hq = (int)(t % H2); t /= H2;
  int c = (int)(t % Cc); int n = (int)(t / Cc);
  int W = 2 * W2;
  const float* xp = x + (long)n * x_img + (long)c * 4 * H2 * W2;
  long base = (long)(2 * hq) * W + 2 * wq;
  float x1 = xp[base] * 0.5f;          // [0::2,0::2]
  float x2 = xp[base + W] * 0.5f;      // [1::2,0::2]
  float x3 = xp[base + 1] * 0.5f;      // [0::2,1::2]
  float x4 = xp[base + W + 1] * 0.5f;  // [1::2,1::2]
  long o = (long)n * o_img + (long)c * H2 * W2 + (long)hq * W2 + wq;
  ll[o] =  x1 + x2 + x3 + x4;
  hl[o] = -x1 - x2 + x3 + x4;
  lh[o] = -x1 + x2 - x3 + x4;
  hh[o] =  x1 - x2 - x3 + x4;
}

__global__ void k_idwt(const float* __restrict__ ll, const float* __restrict__ hl,
                       const float* __restrict__ lh, const float* __restrict__ hh,
                       long ll_img, long h_img, float* __restrict__ out, long o_img,
                       const float* __restrict__ res, long r_img,
                       int Cc, int H2, int W2, int P) {
  long idx = (long)blockIdx.x * blockDim.x + threadIdx.x;
  long total = (long)P * Cc * H2 * W2;
  if (idx >= total) return;
  int wq = (int)(idx % W2); long t = idx / W2;
  int hq = (int)(t % H2); t /= H2;
  int c = (int)(t % Cc); int n = (int)(t / Cc);
  int W = 2 * W2;
  long i = (long)c * H2 * W2 + (long)hq * W2 + wq;
  float L  = ll[(long)n * ll_img + i];
  float h1 = hl[(long)n * h_img + i];
  float l1 = lh[(long)n * h_img + i];
  float h2 = hh[(long)n * h_img + i];
  float a  = (L - h1 - (l1 - h2)) * 0.5f;   // [0::2,0::2]
  float b_ = (L - h1 + (l1 - h2)) * 0.5f;   // [1::2,0::2]
  float c_ = (L + h1 - (l1 + h2)) * 0.5f;   // [0::2,1::2]
  float d_ = (L + h1 + (l1 + h2)) * 0.5f;   // [1::2,1::2]
  long base = (long)(2 * hq) * W + 2 * wq;
  float* op = out + (long)n * o_img + (long)c * 4 * H2 * W2;
  if (res) {
    const float* rp = res + (long)n * r_img + (long)c * 4 * H2 * W2;
    a += rp[base]; b_ += rp[base + W]; c_ += rp[base + 1]; d_ += rp[base + W + 1];
  }
  op[base] = a; op[base + W] = b_; op[base + 1] = c_; op[base + W + 1] = d_;
}

// ---------------------------------------------------------------------------
// LayerNorms, transposes, elementwise helpers
// ---------------------------------------------------------------------------
__global__ void k_ln2d(const float* __restrict__ x, float* __restrict__ y,
                       const float* __restrict__ w, const float* __restrict__ b,
                       int N, int Ch, long HW, float eps) {
  long idx = (long)blockIdx.x * blockDim.x + threadIdx.x;
  long total = (long)N * HW;
  if (idx >= total) return;
  long p = idx % HW; int n = (int)(idx / HW);
  const float* xp = x + (long)n * Ch * HW + p;
  float s = 0.f, sq = 0.f;
  for (int c = 0; c < Ch; c++) { float v = xp[(long)c * HW]; s += v; sq += v * v; }
  float mu = s / Ch;
  float var = sq / Ch - mu * mu;
  float inv = rsqrtf(var + eps);
  float* yp = y + (long)n * Ch * HW + p;
  for (int c = 0; c < Ch; c++) {
    float v = xp[(long)c * HW];
    yp[(long)c * HW] = (v - mu) * inv * w[c] + b[c];
  }
}

__global__ void k_lntok(const float* __restrict__ x, float* __restrict__ y,
                        const float* __restrict__ w, const float* __restrict__ b,
                        long T, int D, float eps) {
  long t = (long)blockIdx.x * blockDim.x + threadIdx.x;
  if (t >= T) return;
  const float* xp = x + t * D;
  float s = 0.f, sq = 0.f;
  for (int d = 0; d < D; d++) { float v = xp[d]; s += v; sq += v * v; }
  float mu = s / D;
  float var = sq / D - mu * mu;
  float inv = rsqrtf(var + eps);
  float* yp = y + t * D;
  for (int d = 0; d < D; d++) yp[d] = (xp[d] - mu) * inv * w[d] + b[d];
}

__global__ void k_nchw2tok(const float* __restrict__ x, float* __restrict__ y,
                           int N, int Ch, long HW) {
  long idx = (long)blockIdx.x * blockDim.x + threadIdx.x;
  long total = (long)N * Ch * HW;
  if (idx >= total) return;
  long p = idx % HW; long t = idx / HW;
  int c = (int)(t % Ch); int n = (int)(t / Ch);
  y[((long)n * HW + p) * Ch + c] = x[idx];
}

__global__ void k_tok2nchw(const float* __restrict__ x, float* __restrict__ y,
                           int N, int Ch, long HW) {
  long idx = (long)blockIdx.x * blockDim.x + threadIdx.x;
  long total = (long)N * Ch * HW;
  if (idx >= total) return;
  long p = idx % HW; long t = idx / HW;
  int c = (int)(t % Ch); int n = (int)(t / Ch);
  y[idx] = x[((long)n * HW + p) * Ch + c];
}

__global__ void k_mulpair(const float* __restrict__ x, float* __restrict__ y,
                          int N, int Chh, long HW) {
  long idx = (long)blockIdx.x * blockDim.x + threadIdx.x;
  long total = (long)N * Chh * HW;
  if (idx >= total) return;
  long p = idx % HW; long t = idx / HW;
  int c = (int)(t % Chh); int n = (int)(t / Chh);
  const float* a = x + ((long)n * 2 * Chh + c) * HW + p;
  const float* b = x + ((long)n * 2 * Chh + Chh + c) * HW + p;
  y[idx] = (*a) * (*b);
}

// ---------------------------------------------------------------------------
// Mamba helpers: xs construction, selective scan, merge, gate
// ---------------------------------------------------------------------------
__global__ void k_build_xs(const float* __restrict__ xc, float* __restrict__ xs,
                           int Bn, int D, int Hh, int Wn) {
  long L = (long)Hh * Wn;
  long idx = (long)blockIdx.x * blockDim.x + threadIdx.x;
  long total = (long)Bn * D * L;
  if (idx >= total) return;
  long l = idx % L; long t = idx / L;
  int d = (int)(t % D); int b = (int)(t / D);
  float v = xc[((long)b * D + d) * L + l];
  int h = (int)(l / Wn), w = (int)(l % Wn);
  long lt = (long)w * Hh + h;
  float* base = xs + (long)b * 4 * D * L + (long)d * L;
  base[0 * D * L + l]            = v;
  base[1 * D * L + lt]           = v;
  base[2 * D * L + (L - 1 - l)]  = v;
  base[3 * D * L + (L - 1 - lt)] = v;
}

// one block per (b,k); 128 threads = DI; 16 states in registers; 32-step
// chunks of dt/u/B/C staged through LDS. ys may alias dts (chunk is staged
// into LDS before being overwritten).
__global__ void __launch_bounds__(128) k_scan(
    const float* __restrict__ dts, const float* __restrict__ xs,
    const float* __restrict__ xdbl, const float* __restrict__ A_log,
    const float* __restrict__ Dsv, float* __restrict__ ys,
    int Kn, int D, int L) {
  __shared__ __align__(16) float dt_s[128 * 32];
  __shared__ __align__(16) float u_s[128 * 32];
  __shared__ float B_s[32 * 16];
  __shared__ float C_s[32 * 16];
  int bk = blockIdx.x;
  int k = bk % Kn;
  int d = threadIdx.x;
  float A[16], h[16];
#pragma unroll
  for (int n = 0; n < 16; n++) {
    A[n] = -__expf(A_log[((long)k * D + d) * 16 + n]);
    h[n] = 0.f;
  }
  float Dd = Dsv[(long)k * D + d];
  const float* dtp = dts + ((long)bk * D + d) * L;
  const float* up  = xs  + ((long)bk * D + d) * L;
  const float* Bp  = xdbl + ((long)bk * 36 + 4)  * L;
  const float* Cp  = xdbl + ((long)bk * 36 + 20) * L;
  float* yp = ys + ((long)bk * D + d) * L;
  for (int l0 = 0; l0 < L; l0 += 32) {
    __syncthreads();
    for (int j = 0; j < 32; j += 4) {
      *(float4*)&dt_s[d * 32 + j] = *(const float4*)&dtp[l0 + j];
      *(float4*)&u_s[d * 32 + j]  = *(const float4*)&up[l0 + j];
    }
    for (int idx = threadIdx.x; idx < 512; idx += 128) {
      int n = idx >> 5, j = idx & 31;
      B_s[j * 16 + n] = Bp[(long)n * L + l0 + j];
      C_s[j * 16 + n] = Cp[(long)n * L + l0 + j];
    }
    __syncthreads();
    for (int j = 0; j < 32; j++) {
      float dt = dt_s[d * 32 + j];
      float u  = u_s[d * 32 + j];
      float du = dt * u;
      float y = 0.f;
#pragma unroll
      for (int n = 0; n < 16; n++) {
        h[n] = __expf(dt * A[n]) * h[n] + du * B_s[j * 16 + n];
        y += h[n] * C_s[j * 16 + n];
      }
      yp[l0 + j] = y + Dd * u;
    }
  }
}

__global__ void k_merge(const float* __restrict__ ys, float* __restrict__ yg,
                        int Bn, int D, int Hh, int Wn) {
  long L = (long)Hh * Wn;
  long idx = (long)blockIdx.x * blockDim.x + threadIdx.x;
  long total = (long)Bn * L * D;
  if (idx >= total) return;
  int d = (int)(idx % D); long t = idx / D;
  long l = t % L; int b = (int)(t / L);
  int h = (int)(l / Wn), w = (int)(l % Wn);
  long lt = (long)w * Hh + h;
  const float* base = ys + (long)b * 4 * D * L + (long)d * L;
  float v = base[0 * D * L + l]
          + base[1 * D * L + lt]
          + base[2 * D * L + (L - 1 - l)]
          + base[3 * D * L + (L - 1 - lt)];
  yg[idx] = v;
}

__global__ void k_gate(float* __restrict__ yg, const float* __restrict__ xz,
                       int Bn, int D, long L) {
  long idx = (long)blockIdx.x * blockDim.x + threadIdx.x;
  long total = (long)Bn * L * D;
  if (idx >= total) return;
  int d = (int)(idx % D); long t = idx / D;
  long l = t % L; int b = (int)(t / L);
  float z = xz[((long)b * 2 * D + D + d) * L + l];
  yg[idx] *= z * (1.f / (1.f + __expf(-z)));
}

// ---------------------------------------------------------------------------
// DFT twiddle matrices (128-point, rfft-style); ortho 1/128 scales applied as GEMM alpha
// ---------------------------------------------------------------------------
__global__ void k_twiddle(float* cosR, float* sinR, float* cosC, float* sinC,
                          float* cosI, float* sinI) {
  int idx = blockIdx.x * blockDim.x + threadIdx.x;
  const float w = 6.283185307179586f / 128.f;
  if (idx < 65 * 128) {          // row fwd: (k,c)
    int k = idx / 128, c = idx % 128; int m = (k * c) & 127;
    cosR[idx] = cosf(w * m); sinR[idx] = sinf(w * m);
  }
  if (idx < 128 * 128) {         // col (symmetric): (a,b)
    int a = idx / 128, bq = idx % 128; int m = (a * bq) & 127;
    cosC[idx] = cosf(w * m); sinC[idx] = sinf(w * m);
  }
  if (idx < 128 * 65) {          // row inv: (c,k), Hermitian fold weights
    int c = idx / 65, k = idx % 65; int m = (k * c) & 127;
    float wk = (k == 0 || k == 64) ? 1.f : 2.f;
    cosI[idx] = wk * cosf(w * m); sinI[idx] = wk * sinf(w * m);
  }
}

// ---------------------------------------------------------------------------
// Host-side helpers
// ---------------------------------------------------------------------------
static inline dim3 g1d(long total) { return dim3((unsigned)((total + 255) / 256)); }

static void gemm(hipStream_t st, int M, int N, int K, int z,
                 const float* A, long as_m, long as_k, long bA,
                 const float* B, long bs_k, long bs_n, long bB,
                 float* C, long cs_m, long cs_n, long bC,
                 const float* bias, long bBias, float alpha, int act, int acc) {
  GemmP p{};
  p.A = A; p.B = B; p.C = C; p.bias = bias;
  p.as_m = as_m; p.as_k = as_k; p.batchA = bA;
  p.bs_k = bs_k; p.bs_n = bs_n; p.batchB = bB;
  p.cs_m = cs_m; p.cs_n = cs_n; p.batchC = bC;
  p.batchBias = bBias; p.M = M; p.N = N; p.K = K;
  p.alpha = alpha; p.act = act; p.acc = acc;
  p.has_bias = (bias != nullptr); p.im2col = 0;
  dim3 g((N + 63) / 64, (M + 63) / 64, z);
  k_gemm<<<g, dim3(128), 0, st>>>(p);
}

static void conv_gemm(hipStream_t st, int Cout, int Cin, int H, int W, int ks, int z,
                      const float* Wt, const float* X, long imB,
                      float* C, long cs_m, long cs_n, long bC,
                      const float* bias, int act, int acc) {
  int K = Cin * ks * ks, N = H * W;
  GemmP p{};
  p.A = Wt; p.as_m = K; p.as_k = 1; p.batchA = 0;
  p.Xim = X; p.imBatch = imB; p.im2col = 1;
  p.Cin = Cin; p.Hh = H; p.Ww = W; p.kh = ks; p.kw = ks;
  p.C = C; p.cs_m = cs_m; p.cs_n = cs_n; p.batchC = bC;
  p.bias = bias; p.has_bias = (bias != nullptr); p.batchBias = 0;
  p.M = Cout; p.N = N; p.K = K; p.alpha = 1.f; p.act = act; p.acc = acc;
  dim3 g((N + 63) / 64, (Cout + 63) / 64, z);
  k_gemm<<<g, dim3(128), 0, st>>>(p);
}

struct FftTw { const float *cosR, *sinR, *cosC, *sinC, *cosI, *sinI; };

// FFC block: out = in + conv3(h + irfft2(relu(spec(rfft2(h))))) ; h=relu(conv3(in))
static void run_ffc(hipStream_t st, const float* in, float* out,
                    const float* c1w, const float* c1b,
                    const float* specw, const float* specb,
                    const float* c2w, const float* c2b,
                    float* h, float* Re1, float* Im1, float* fcat, float* fcat2,
                    float* G1re, float* G1im, const FftTw& tw) {
  const long CHW = 64L * 128 * 128;           // per-image
  const int T = 2 * 64 * 128;                 // 16384 rows (b,ch,r)
  // h = relu(conv3x3(in))
  conv_gemm(st, 64, 64, 128, 128, 3, 2, c1w, in, CHW, h, 128L * 128, 1, CHW, c1b, 1, 0);
  // forward row DFT: Re1 = cosR@h, Im1 = -sinR@h   (t,65)
  gemm(st, 65, T, 128, 1, tw.cosR, 128, 1, 0, h, 1, 128, 0, Re1, 1, 65, 0, nullptr, 0,  1.f, 0, 0);
  gemm(st, 65, T, 128, 1, tw.sinR, 128, 1, 0, h, 1, 128, 0, Im1, 1, 65, 0, nullptr, 0, -1.f, 0, 0);
  // forward col DFT per batch (z = 64 channel images); ortho scale 1/128 here
  for (int b = 0; b < 2; b++) {
    const float* Rb = Re1 + (long)b * 64 * 128 * 65;
    const float* Ib = Im1 + (long)b * 64 * 128 * 65;
    float* Fre = fcat + (long)b * 128 * 8320;
    float* Fim = Fre + 64L * 8320;
    long ib = 128L * 65;
    gemm(st, 128, 65, 128, 64, tw.cosC, 128, 1, 0, Rb, 65, 1, ib, Fre, 65, 1, 8320, nullptr, 0,  1.f / 128.f, 0, 0);
    gemm(st, 128, 65, 128, 64, tw.sinC, 128, 1, 0, Ib, 65, 1, ib, Fre, 65, 1, 8320, nullptr, 0,  1.f / 128.f, 0, 1);
    gemm(st, 128, 65, 128, 64, tw.cosC, 128, 1, 0, Ib, 65, 1, ib, Fim, 65, 1, 8320, nullptr, 0,  1.f / 128.f, 0, 0);
    gemm(st, 128, 65, 128, 64, tw.sinC, 128, 1, 0, Rb, 65, 1, ib, Fim, 65, 1, 8320, nullptr, 0, -1.f / 128.f, 0, 1);
  }
  // spectral 1x1 + relu over 2C channels
  gemm(st, 128, 8320, 128, 2, specw, 128, 1, 0, fcat, 8320, 1, 128L * 8320,
       fcat2, 8320, 1, 128L * 8320, specb, 0, 1.f, 1, 0);
  // inverse col DFT (e^{+i})
  for (int b = 0; b < 2; b++) {
    const float* fr = fcat2 + (long)b * 128 * 8320;
    const float* fi = fr + 64L * 8320;
    float* Gr = G1re + (long)b * 64 * 128 * 65;
    float* Gi = G1im + (long)b * 64 * 128 * 65;
    long ob = 128L * 65;
    gemm(st, 128, 65, 128, 64, tw.cosC, 128, 1, 0, fr, 65, 1, 8320, Gr, 65, 1, ob, nullptr, 0,  1.f, 0, 0);
    gemm(st, 128, 65, 128, 64, tw.sinC, 128, 1, 0, fi, 65, 1, 8320, Gr, 65, 1, ob, nullptr, 0, -1.f, 0, 1);
    gemm(st, 128, 65, 128, 64, tw.cosC, 128, 1, 0, fi, 65, 1, 8320, Gi, 65, 1, ob, nullptr, 0,  1.f, 0, 0);
    gemm(st, 128, 65, 128, 64, tw.sinC, 128, 1, 0, fr, 65, 1, 8320, Gi, 65, 1, ob, nullptr, 0,  1.f, 0, 1);
  }
  // inverse row DFT with Hermitian fold weights, accumulate into h (h += hs)
  gemm(st, 128, T, 65, 1, tw.cosI, 65, 1, 0, G1re, 1, 65, 0, h, 1, 128, 0, nullptr, 0,  1.f / 128.f, 0, 1);
  gemm(st, 128, T, 65, 1, tw.sinI, 65, 1, 0, G1im, 1, 65, 0, h, 1, 128, 0, nullptr, 0, -1.f / 128.f, 0, 1);
  // out = in + conv3x3(h)
  hipMemcpyAsync(out, in, 2 * CHW * sizeof(float), hipMemcpyDeviceToDevice, st);
  conv_gemm(st, 64, 64, 128, 128, 3, 2, c2w, h, CHW, out, 128L * 128, 1, CHW, c2b, 0, 1);
}

// wtconv2d: depthwise 5x5 base + 3-level Haar pyramid of depthwise 5x5 convs
struct WtBufs { float *base, *cat0a, *cat0b, *cat1a, *cat1b, *cat2a, *cat2b, *rec16, *rec32; };

static void wtconv(hipStream_t st, const float* in, float* out,
                   const float* base_w, const float* base_b, const float* base_scale,
                   const float* wav_w, const float* wav_b, const float* wav_scale,
                   const WtBufs& wb) {
  const long S = 64L * 64 * 64;  // image stride at 64x64 (C*H*W)
  // base conv
  k_dwconv<<<g1d(2 * S), 256, 0, st>>>(in, base_w, base_b, base_scale, wb.base,
                                       2, 64, 64, 64, 5, S, S, 0);
  // level 0: 64 -> 32
  k_dwt<<<g1d(2L * 64 * 32 * 32), 256, 0, st>>>(in, S, 64, 32, 32,
      wb.cat0a, wb.cat0a + 64L * 1024, wb.cat0a + 128L * 1024, wb.cat0a + 192L * 1024,
      256L * 1024, 2);
  k_dwconv<<<g1d(2L * 256 * 32 * 32), 256, 0, st>>>(wb.cat0a, wav_w, wav_b, wav_scale,
      wb.cat0b, 2, 256, 32, 32, 5, 256L * 1024, 256L * 1024, 0);
  // level 1: 32 -> 16
  k_dwt<<<g1d(2L * 64 * 16 * 16), 256, 0, st>>>(wb.cat0b, 256L * 1024, 64, 16, 16,
      wb.cat1a, wb.cat1a + 64L * 256, wb.cat1a + 128L * 256, wb.cat1a + 192L * 256,
      256L * 256, 2);
  k_dwconv<<<g1d(2L * 256 * 16 * 16), 256, 0, st>>>(wb.cat1a, wav_w + 6400, wav_b + 256,
      wav_scale + 256, wb.cat1b, 2, 256, 16, 16, 5, 256L * 256, 256L * 256, 0);
  // level 2: 16 -> 8
  k_dwt<<<g1d(2L * 64 * 8 * 8), 256, 0, st>>>(wb.cat1b, 256L * 256, 64, 8, 8,
      wb.cat2a, wb.cat2a + 64L * 64, wb.cat2a + 128L * 64, wb.cat2a + 192L * 64,
      256L * 64, 2);
  k_dwconv<<<g1d(2L * 256 * 8 * 8), 256, 0, st>>>(wb.cat2a, wav_w + 12800, wav_b + 512,
      wav_scale + 512, wb.cat2b, 2, 256, 8, 8, 5, 256L * 64, 256L * 64, 0);
  // idwt chain back up; final level adds base
  k_idwt<<<g1d(2L * 64 * 8 * 8), 256, 0, st>>>(wb.cat2b, wb.cat2b + 4096,
      wb.cat2b + 8192, wb.cat2b + 12288, 16384L, 16384L, wb.rec16, 16384L,
      nullptr, 0, 64, 8, 8, 2);
  k_idwt<<<g1d(2L * 64 * 16 * 16), 256, 0, st>>>(wb.rec16, wb.cat1b + 16384,
      wb.cat1b + 32768, wb.cat1b + 49152, 16384L, 65536L, wb.rec32, 65536L,
      nullptr, 0, 64, 16, 16, 2);
  k_idwt<<<g1d(2L * 64 * 32 * 32), 256, 0, st>>>(wb.rec32, wb.cat0b + 65536,
      wb.cat0b + 131072, wb.cat0b + 196608, 65536L, 262144L, out, 262144L,
      wb.base, 262144L, 64, 32, 32, 2);
}

// ---------------------------------------------------------------------------
// Input flattening order (jax.tree_util: dicts sorted by key, lists in order).
// ---------------------------------------------------------------------------
enum {
  I_DIAG_W = 0, I_FUSE_W, I_HH_DW_B, I_HH_DW_W, I_HH_PW_B, I_HH_PW_W,
  I_FF_B1, I_FF_B2, I_FF_LN_B, I_FF_LN_W, I_FF_W1, I_FF_W2,
  I_A_LOG, I_DSV, I_CONV_B, I_CONV_W, I_DT_B, I_DT_W, I_IN_PROJ, I_ONB, I_ONW,
  I_OUT_PROJ, I_X_PROJ,
  I_HOR_W,
  I_L0_BASE_B, I_L0_BASE_S, I_L0_BASE_W, I_L0_WAV_B, I_L0_WAV_S, I_L0_WAV_W,
  I_L1_BASE_B, I_L1_BASE_S, I_L1_BASE_W, I_L1_WAV_B, I_L1_WAV_S, I_L1_WAV_W,
  I_NORM2_B, I_NORM2_W,
  I_CIM_C1_B, I_CIM_C1_W, I_CIM_C2_B, I_CIM_C2_W,
  I_CF_C1_B, I_CF_C1_W, I_CF_C2_B, I_CF_C2_W, I_CF_SPEC_B, I_CF_SPEC_W,
  I_CIM_LN_B, I_CIM_LN_W,
  I_SIM_C1_B, I_SIM_C1_W, I_SIM_C5_B, I_SIM_C5_W,
  I_SF_C1_B, I_SF_C1_W, I_SF_C2_B, I_SF_C2_W, I_SF_SPEC_B, I_SF_SPEC_W,
  I_SIM_LN_B, I_SIM_LN_W,
  I_VER_W, I_X
};

extern "C" void kernel_launch(void* const* d_in, const int* in_sizes, int n_in,
                              void* d_out, int out_size, void* d_ws, size_t ws_size,
                              hipStream_t stream) {
  (void)in_sizes; (void)n_in; (void)out_size; (void)ws_size;
  hipStream_t st = stream;
  auto F = [&](int i) -> const float* { return (const float*)d_in[i]; };
  const float* X = F(I_X);
  float* OUT = (float*)d_out;

  // ----- workspace bump arena ----------------------------------------------
  char* wsb = (char*)d_ws;
  size_t off = 0;
  auto alloc = [&](size_t nfloat) -> float* {
    float* p = (float*)(wsb + off);
    off += ((nfloat * 4 + 255) / 256) * 256;
    return p;
  };
  const long PS = 64L * 4096;           // plane-set stride at 64x64 (C*H2*W2)
  const long PCHW = 64L * 16384;        // plane-set stride at 128x128

  // persistent region
  float* lle_out = alloc(2 * PS);
  float* ehigh   = alloc(6 * PS);
  float* xout    = alloc(2 * PCHW);
  float* cosR = alloc(65 * 128); float* sinR = alloc(65 * 128);
  float* cosC = alloc(128 * 128); float* sinC = alloc(128 * 128);
  float* cosI = alloc(128 * 65); float* sinI = alloc(128 * 65);
  size_t mark = off;
  FftTw tw{ cosR, sinR, cosC, sinC, cosI, sinI };
  k_twiddle<<<g1d(128 * 128), 256, 0, st>>>(cosR, sinR, cosC, sinC, cosI, sinI);

  // ======================= Phase A: wavelet front-end ======================
  float* dwt1 = alloc(8 * PS);          // [ll(2) | hl(2) | lh(2) | hh(2)]
  float* lle_tmp = alloc(2 * PS);
  WtBufs wb;
  wb.base  = alloc(2 * PS);
  wb.cat0a = alloc(2L * 256 * 1024); wb.cat0b = alloc(2L * 256 * 1024);
  wb.cat1a = alloc(2L * 256 * 256);  wb.cat1b = alloc(2L * 256 * 256);
  wb.cat2a = alloc(2L * 256 * 64);   wb.cat2b = alloc(2L * 256 * 64);
  wb.rec16 = alloc(2L * 64 * 256);   wb.rec32 = alloc(2L * 64 * 1024);
  float* hh3a = alloc(6 * PS);
  float* ehigh_cat = alloc(6L * 128 * 4096);

  // haar_dwt(x)
  k_dwt<<<g1d(2 * PS), 256, 0, st>>>(X, PCHW, 64, 64, 64,
      dwt1, dwt1 + 2 * PS, dwt1 + 4 * PS, dwt1 + 6 * PS, PS, 2);
  // llenhance x2
  wtconv(st, dwt1, lle_tmp, F(I_L0_BASE_W), F(I_L0_BASE_B), F(I_L0_BASE_S),
         F(I_L0_WAV_W), F(I_L0_WAV_B), F(I_L0_WAV_S), wb);
  wtconv(st, lle_tmp, lle_out, F(I_L1_BASE_W), F(I_L1_BASE_B), F(I_L1_BASE_S),
         F(I_L1_WAV_W), F(I_L1_WAV_B), F(I_L1_WAV_S), wb);
  // hh branch: depthwise 3x3 on [hl;lh;hh] (6 images) then pointwise C->C into
  // e_high_cat channels [0,64)
  k_dwconv<<<g1d(6 * PS), 256, 0, st>>>(dwt1 + 2 * PS, F(I_HH_DW_W), F(I_HH_DW_B),
      nullptr, hh3a, 6, 64, 64, 64, 3, PS, PS, 0);
  gemm(st, 64, 4096, 64, 6, F(I_HH_PW_W), 64, 1, 0, hh3a, 4096, 1, PS,
       ehigh_cat, 4096, 1, 2 * PS, F(I_HH_PW_B), 0, 1.f, 0, 0);
  // hor/ver/diag dense 3x3 on enhanced ll into channels [64,128)
  const float* dirw[3] = { F(I_HOR_W), F(I_VER_W), F(I_DIAG_W) };
  for (int i = 0; i < 3; i++)
    conv_gemm(st, 64, 64, 64, 64, 3, 2, dirw[i], lle_out, PS,
              ehigh_cat + (long)i * 2 * (2 * PS) + PS, 4096, 1, 2 * PS,
              nullptr, 0, 0);
  // fuse 1x1 2C->C, then ln2d (eps 1e-5)
  gemm(st, 64, 4096, 128, 6, F(I_FUSE_W), 128, 1, 0, ehigh_cat, 4096, 1, 2 * PS,
       ehigh, 4096, 1, PS, nullptr, 0, 1.f, 0, 0);
  k_ln2d<<<g1d(6L * 4096), 256, 0, st>>>(ehigh, ehigh, F(I_NORM2_W), F(I_NORM2_B),
      6, 64, 4096, 1e-5f);

  // ======================= Phase B: hhmamba (SS2D + FF) ====================
  off = mark;
  const int Bm = 6, DIc = 128, Lm = 4096;
  const long TOK = (long)Bm * Lm;      // 24576 tokens
  float* xz    = alloc((long)Bm * 256 * Lm);
  float* xconv = alloc((long)Bm * DIc * Lm);
  float* xs_b  = alloc((long)Bm * 4 * DIc * Lm);
  float* xdbl  = alloc((long)Bm * 4 * 36 * Lm);
  float* dts_b = alloc((long)Bm * 4 * DIc * Lm);  // also reused as ys
  float* t1 = alloc(TOK * 64);
  float* t2 = alloc(TOK * 64);
  float* ys = dts_b;                               // scan writes in place
  float* yg = xdbl;                                // merged y (aliases x_dbl)
  float* g1 = xs_b;                                // FFN hidden (aliases xs)

  // in_proj: tokens(C=64) -> 2*DI, output channel-major (B,256,L)
  gemm(st, 256, Lm, 64, Bm, F(I_IN_PROJ), 1, 256, 0, ehigh, Lm, 1, PS,
       xz, Lm, 1, 256L * Lm, nullptr, 0, 1.f, 0, 0);
  // depthwise 3x3 + silu on x_ part
  k_dwconv<<<g1d((long)Bm * DIc * Lm), 256, 0, st>>>(xz, F(I_CONV_W), F(I_CONV_B),
      nullptr, xconv, Bm, DIc, 64, 64, 3, 256L * Lm, (long)DIc * Lm, 3);
  // 4 scan orderings
  k_build_xs<<<g1d((long)Bm * DIc * Lm), 256, 0, st>>>(xconv, xs_b, Bm, DIc, 64, 64);
  // x_proj (per direction) and dt projection (+softplus)
  for (int b = 0; b < Bm; b++) {
    gemm(st, 36, Lm, DIc, 4, F(I_X_PROJ), 1, 36, (long)DIc * 36,
         xs_b + (long)b * 4 * DIc * Lm, Lm, 1, (long)DIc * Lm,
         xdbl + (long)b * 4 * 36 * Lm, Lm, 1, 36L * Lm, nullptr, 0, 1.f, 0, 0);
    gemm(st, DIc, Lm, 4, 4, F(I_DT_W), 1, DIc, 4L * DIc,
         xdbl + (long)b * 4 * 36 * Lm, Lm, 1, 36L * Lm,
         dts_b + (long)b * 4 * DIc * Lm, Lm, 1, (long)DIc * Lm,
         F(I_DT_B), DIc, 1.f, 4 /*softplus*/, 0);
  }
  // selective scan (sequential over L, per (b,k) block)
  k_scan<<<dim3(Bm * 4), dim3(128), 0, st>>>(dts_b, xs_b, xdbl, F(I_A_LOG),
      F(I_DSV), ys, 4, DIc, Lm);
  // merge 4 directions into token-major y, LN, gate with silu(z)
  k_merge<<<g1d(TOK * DIc), 256, 0, st>>>(ys, yg, Bm, DIc, 64, 64);
  k_lntok<<<g1d(TOK), 256, 0, st>>>(yg, yg, F(I_ONW), F(I_ONB), TOK, DIc, 1e-5f);
  k_gate<<<g1d(TOK * DIc), 256, 0, st>>>(yg, xz, Bm, DIc, Lm);
  // residual tokens; out_proj accumulates into them
  k_nchw2tok<<<g1d(6 * PS), 256, 0, st>>>(ehigh, t1, 6, 64, 4096);
  gemm(st, 64, (int)TOK, DIc, 1, F(I_OUT_PROJ), 1, 64, 0, yg, 1, DIc, 0,
       t1, 1, 64, 0, nullptr, 0, 1.f, 0, 1);
  // pre-norm FFN with residual
  k_lntok<<<g1d(TOK), 256, 0, st>>>(t1, t2, F(I_FF_LN_W), F(I_FF_LN_B), TOK, 64, 1e-5f);
  gemm(st, 256, (int)TOK, 64, 1, F(I_FF_W1), 1, 256, 0, t2, 1, 64, 0,
       g1, 1, 256, 0, F(I_FF_B1), 0, 1.f, 2 /*gelu*/, 0);
  gemm(st, 64, (int)TOK, 256, 1, F(I_FF_W2), 1, 64, 0, g1, 1, 256, 0,
       t1, 1, 64, 0, F(I_FF_B2), 0, 1.f, 0, 1);
  k_tok2nchw<<<g1d(6 * PS), 256, 0, st>>>(t1, ehigh, 6, 64, 4096);

  // inverse Haar with residual x -> x_out
  k_idwt<<<g1d(2 * PS), 256, 0, st>>>(lle_out, ehigh, ehigh + 2 * PS,
      ehigh + 4 * PS, PS, PS, xout, PCHW, X, PCHW, 64, 64, 64, 2);

  // ======================= Phase C: posenhance light_block =================
  off = mark;
  float* ybuf  = alloc(2 * PCHW);
  float* h     = alloc(2 * PCHW);
  float* Re1   = alloc(16384L * 65);
  float* Im1   = alloc(16384L * 65);
  float* fcat  = alloc(2L * 128 * 8320);
  float* fcat2 = alloc(2L * 128 * 8320);
  float* G1re  = alloc(16384L * 65);
  float* G1im  = alloc(16384L * 65);
  float* ffco  = alloc(2 * PCHW);
  float* c5out = alloc(2 * PCHW);
  float* prod  = alloc(2L * 32 * 16384);
  float* ybuf2 = alloc(2 * PCHW);
  float* zbuf  = alloc(2 * PCHW);
  float* zcat  = alloc(2L * 256 * 16384);
  float* zprod = alloc(2L * 128 * 16384);

  // sim path
  k_ln2d<<<g1d(2L * 16384), 256, 0, st>>>(xout, ybuf, F(I_SIM_LN_W), F(I_SIM_LN_B),
      2, 64, 16384, 1e-6f);
  run_ffc(st, ybuf, ffco, F(I_SF_C1_W), F(I_SF_C1_B), F(I_SF_SPEC_W), F(I_SF_SPEC_B),
          F(I_SF_C2_W), F(I_SF_C2_B), h, Re1, Im1, fcat, fcat2, G1re, G1im, tw);
  conv_gemm(st, 64, 64, 128, 128, 5, 2, F(I_SIM_C5_W), ffco, PCHW,
            c5out, 16384, 1, PCHW, F(I_SIM_C5_B), 0, 0);
  k_mulpair<<<g1d(2L * 32 * 16384), 256, 0, st>>>(c5out, prod, 2, 32, 16384);
  hipMemcpyAsync(ybuf2, xout, 2 * PCHW * sizeof(float), hipMemcpyDeviceToDevice, st);
  gemm(st, 64, 16384, 32, 2, F(I_SIM_C1_W), 32, 1, 0, prod, 16384, 1, 32L * 16384,
       ybuf2, 16384, 1, PCHW, F(I_SIM_C1_B), 0, 1.f, 0, 1);

  // cim path
  k_ln2d<<<g1d(2L * 16384), 256, 0, st>>>(ybuf2, zbuf, F(I_CIM_LN_W), F(I_CIM_LN_B),
      2, 64, 16384, 1e-6f);
  run_ffc(st, zbuf, ffco, F(I_CF_C1_W), F(I_CF_C1_B), F(I_CF_SPEC_W), F(I_CF_SPEC_B),
          F(I_CF_C2_W), F(I_CF_C2_B), h, Re1, Im1, fcat, fcat2, G1re, G1im, tw);
  gemm(st, 256, 16384, 64, 2, F(I_CIM_C1_W), 64, 1, 0, ffco, 16384, 1, PCHW,
       zcat, 16384, 1, 256L * 16384, F(I_CIM_C1_B), 0, 1.f, 0, 0);
  k_mulpair<<<g1d(2L * 128 * 16384), 256, 0, st>>>(zcat, zprod, 2, 128, 16384);
  hipMemcpyAsync(OUT, ybuf2, 2 * PCHW * sizeof(float), hipMemcpyDeviceToDevice, st);
  gemm(st, 64, 16384, 128, 2, F(I_CIM_C2_W), 128, 1, 0, zprod, 16384, 1, 128L * 16384,
       OUT, 16384, 1, PCHW, F(I_CIM_C2_B), 0, 1.f, 0, 1);
}
